// FeaturePropagationModule_85641647882660
// MI455X (gfx1250) — matrix operand order repaired and checked by reading, not compiled, not run on hardware
//
#include <hip/hip_runtime.h>
#include <hip/hip_bf16.h>
#include <math.h>

typedef __attribute__((ext_vector_type(16))) _Float16 v16h;
typedef __attribute__((ext_vector_type(8)))  float    v8f;

// ---------------- degree / norm kernels ----------------

__global__ void k_init_deg(float* __restrict__ deg, int N) {
    int i = blockIdx.x * blockDim.x + threadIdx.x;
    if (i < N) deg[i] = 1.0f;               // self-loop contributes 1
}

__global__ void k_count_deg(const int* __restrict__ col, float* __restrict__ deg, long long E) {
    long long e = (long long)blockIdx.x * blockDim.x + threadIdx.x;
    if (e < E) atomicAdd(&deg[col[e]], 1.0f);
}

__global__ void k_rsqrt(const float* __restrict__ deg, float* __restrict__ dis, int N) {
    int i = blockIdx.x * blockDim.x + threadIdx.x;
    if (i < N) dis[i] = rsqrtf(deg[i]);
}

__global__ void k_norm(const int* __restrict__ row, const int* __restrict__ col,
                       const float* __restrict__ dis, float* __restrict__ nrm, long long E) {
    long long e = (long long)blockIdx.x * blockDim.x + threadIdx.x;
    if (e < E) nrm[e] = dis[row[e]] * dis[col[e]];
}

// ---------------- WMMA GEMM: C[M x Fout] = A[M x K] * W[K x Fout] ----------------
// One wave owns one 16-row m-strip and all NT column tiles: A fragment loaded
// once per k-step, NT WMMAs issued against pre-swizzled B fragments in LDS.
// LDS holds W in B-fragment order: frag(kblk, nt) -> lane*16 halves contiguous,
// so each lane fetches its fragment as two ds_load_b128.

template <int NT>
__global__ void k_gemm_wmma(const float* __restrict__ A, const float* __restrict__ W,
                            float* __restrict__ C, int M, int K, int Fout, int MT) {
    // chunk of up to 128 K-rows: (128/32)*NT frags * 512 halves
    __shared__ _Float16 lw[4 * NT * 512];
    const int tid   = threadIdx.x;
    const int wave  = tid >> 5;
    const int lane  = tid & 31;
    const int hhalf = lane >> 4;     // 0: lanes 0-15, 1: lanes 16-31
    const int mr    = lane & 15;

    const int mTile = blockIdx.x * 8 + wave;
    const bool active = mTile < MT;              // wave-uniform

    const float* arow = A + (size_t)((active ? mTile : 0) * 16 + mr) * K;
    v8f acc[NT];
    #pragma unroll
    for (int nt = 0; nt < NT; ++nt) acc[nt] = (v8f){};

    const v16h* lwv = (const v16h*)lw;

    for (int kc = 0; kc < K; kc += 128) {
        const int kLen = (K - kc < 128) ? (K - kc) : 128;   // multiple of 32
        // ---- stage W chunk into LDS in B-fragment order ----
        const int total = (kLen >> 5) * NT * 512;
        for (int idx = tid; idx < total; idx += 256) {
            int fragId = idx >> 9;           // (kblk * NT + nt)
            int r      = idx & 511;
            int ln     = r >> 4;             // fragment lane
            int j      = r & 15;             // half index within lane
            int kblk   = fragId / NT;
            int nt     = fragId - kblk * NT;
            int k = kc + kblk * 32 + ((ln & 16) >> 1) + (j & 7) + ((j >> 3) << 4);   // +8 for lanes 16-31; j>=8 -> +16
            int n = nt * 16 + (ln & 15);
            float v = (n < Fout) ? W[(size_t)k * Fout + n] : 0.0f;
            lw[idx] = (_Float16)v;
        }
        __syncthreads();

        if (active) {
            for (int k0 = 0; k0 < kLen; k0 += 32) {
                // A fragment (16x32 f16), loaded once, reused for NT WMMAs
                v16h a;
                const float4* p = (const float4*)(arow + kc + k0 + hhalf * 8);
                float4 f0 = p[0];
                float4 f1 = p[1];
                const float4* q = (const float4*)(arow + kc + k0 + hhalf * 8 + 16);
                float4 f2 = q[0];
                float4 f3 = q[1];
                a[0]=(_Float16)f0.x; a[1]=(_Float16)f0.y; a[2]=(_Float16)f0.z; a[3]=(_Float16)f0.w;
                a[4]=(_Float16)f1.x; a[5]=(_Float16)f1.y; a[6]=(_Float16)f1.z; a[7]=(_Float16)f1.w;
                a[8]=(_Float16)f2.x; a[9]=(_Float16)f2.y; a[10]=(_Float16)f2.z; a[11]=(_Float16)f2.w;
                a[12]=(_Float16)f3.x; a[13]=(_Float16)f3.y; a[14]=(_Float16)f3.z; a[15]=(_Float16)f3.w;

                const int fragBase = (k0 >> 5) * NT;
                #pragma unroll
                for (int nt = 0; nt < NT; ++nt) {
                    v16h b = lwv[(fragBase + nt) * 32 + lane];   // 2x ds_load_b128
                    acc[nt] = __builtin_amdgcn_wmma_f32_16x16x32_f16(
                                  false, a, false, b, (short)0, acc[nt], false, false);
                }
            }
        }
        __syncthreads();
    }

    if (active) {
        const int mbase = mTile * 16 + hhalf * 8;
        #pragma unroll
        for (int nt = 0; nt < NT; ++nt) {
            const int n = nt * 16 + mr;
            if (n < Fout) {
                #pragma unroll
                for (int r = 0; r < 8; ++r)
                    C[(size_t)(mbase + r) * Fout + n] = acc[nt][r];
            }
        }
    }
}

// ---------------- aggregation ----------------

// agg[i][f] = xw[i][f] * dis[i]^2 (self-loop) + b[f]
__global__ void k_init_agg(const float* __restrict__ xw, const float* __restrict__ dis,
                           const float* __restrict__ bias, float* __restrict__ agg,
                           int N, int F) {
    long long t = (long long)blockIdx.x * blockDim.x + threadIdx.x;
    if (t >= (long long)N * F) return;
    int i = (int)(t / F);
    int f = (int)(t - (long long)i * F);
    float d = dis[i];
    agg[t] = xw[t] * d * d + bias[f];
}

// agg[col[e]][f] += xw[row[e]][f] * norm[e]
__global__ void k_scatter(const float* __restrict__ xw, const float* __restrict__ nrm,
                          const int* __restrict__ row, const int* __restrict__ col,
                          float* __restrict__ agg, long long E, int F) {
    long long t = (long long)blockIdx.x * blockDim.x + threadIdx.x;
    if (t >= E * (long long)F) return;
    long long e = t / F;
    int f = (int)(t - e * F);
    float v = xw[(size_t)row[e] * F + f] * nrm[e];
    atomicAdd(&agg[(size_t)col[e] * F + f], v);
}

__global__ void k_relu(const float* __restrict__ in, float* __restrict__ out, long long n) {
    long long t = (long long)blockIdx.x * blockDim.x + threadIdx.x;
    if (t < n) out[t] = fmaxf(in[t], 0.0f);
}

__global__ void k_log_softmax(const float* __restrict__ in, float* __restrict__ out,
                              int N, int C) {
    int i = blockIdx.x * blockDim.x + threadIdx.x;
    if (i >= N) return;
    const float* r = in + (size_t)i * C;
    float mx = r[0];
    for (int j = 1; j < C; ++j) mx = fmaxf(mx, r[j]);
    float s = 0.0f;
    for (int j = 0; j < C; ++j) s += expf(r[j] - mx);
    float lse = logf(s) + mx;
    float* o = out + (size_t)i * C;
    for (int j = 0; j < C; ++j) o[j] = r[j] - lse;
}

// ---------------- launcher ----------------

static inline unsigned gblocks(long long n, int bs) {
    return (unsigned)((n + bs - 1) / bs);
}

static void launch_gemm(const float* A, const float* W, float* C,
                        int M, int K, int Fout, hipStream_t stream) {
    const int MT = (M + 15) / 16;
    const int NT = (Fout + 15) / 16;
    const unsigned blocks = gblocks(MT, 8);
    switch (NT) {
        case 3: k_gemm_wmma<3><<<blocks, 256, 0, stream>>>(A, W, C, M, K, Fout, MT); break;
        case 4: k_gemm_wmma<4><<<blocks, 256, 0, stream>>>(A, W, C, M, K, Fout, MT); break;
        case 8: k_gemm_wmma<8><<<blocks, 256, 0, stream>>>(A, W, C, M, K, Fout, MT); break;
        case 1: k_gemm_wmma<1><<<blocks, 256, 0, stream>>>(A, W, C, M, K, Fout, MT); break;
        case 2: k_gemm_wmma<2><<<blocks, 256, 0, stream>>>(A, W, C, M, K, Fout, MT); break;
        case 5: k_gemm_wmma<5><<<blocks, 256, 0, stream>>>(A, W, C, M, K, Fout, MT); break;
        case 6: k_gemm_wmma<6><<<blocks, 256, 0, stream>>>(A, W, C, M, K, Fout, MT); break;
        default: k_gemm_wmma<7><<<blocks, 256, 0, stream>>>(A, W, C, M, K, Fout, MT); break;
    }
}

extern "C" void kernel_launch(void* const* d_in, const int* in_sizes, int n_in,
                              void* d_out, int out_size, void* d_ws, size_t ws_size,
                              hipStream_t stream) {
    const float* x  = (const float*)d_in[0];
    const int*   ei = (const int*)  d_in[1];
    const float* W1 = (const float*)d_in[2];
    const float* b1 = (const float*)d_in[3];
    const float* W2 = (const float*)d_in[4];
    const float* b2 = (const float*)d_in[5];
    const float* W3 = (const float*)d_in[6];
    const float* b3 = (const float*)d_in[7];
    float* out = (float*)d_out;

    const int Fin = 512;
    const int N   = in_sizes[0] / Fin;          // 100000
    const long long E = in_sizes[1] / 2;        // 1600000
    const int F1  = in_sizes[2] / Fin;          // 64
    const int F2  = in_sizes[4] / F1;           // 128
    const int CC  = in_sizes[6] / F2;           // 40
    const int* row = ei;
    const int* col = ei + E;

    float* ws  = (float*)d_ws;
    float* deg = ws;                            // N
    float* dis = deg + N;                       // N
    float* nrm = dis + N;                       // E
    float* xw  = nrm + E;                       // N*F2 max
    float* agg = xw  + (size_t)N * F2;          // N*F2 max
    float* act = agg + (size_t)N * F2;          // N*F2 max

    const int BS = 256;

    // degree + normalization
    k_init_deg <<<gblocks(N, BS), BS, 0, stream>>>(deg, N);
    k_count_deg<<<gblocks(E, BS), BS, 0, stream>>>(col, deg, E);
    k_rsqrt    <<<gblocks(N, BS), BS, 0, stream>>>(deg, dis, N);
    k_norm     <<<gblocks(E, BS), BS, 0, stream>>>(row, col, dis, nrm, E);

    // ---- layer 1: x[N x 512] * W1 -> 64, scatter, relu ----
    launch_gemm(x, W1, xw, N, Fin, F1, stream);
    k_init_agg<<<gblocks((long long)N * F1, BS), BS, 0, stream>>>(xw, dis, b1, agg, N, F1);
    k_scatter <<<gblocks(E * F1, BS), BS, 0, stream>>>(xw, nrm, row, col, agg, E, F1);
    k_relu    <<<gblocks((long long)N * F1, BS), BS, 0, stream>>>(agg, act, (long long)N * F1);

    // ---- layer 2: act[N x 64] * W2 -> 128, scatter, relu ----
    launch_gemm(act, W2, xw, N, F1, F2, stream);
    k_init_agg<<<gblocks((long long)N * F2, BS), BS, 0, stream>>>(xw, dis, b2, agg, N, F2);
    k_scatter <<<gblocks(E * F2, BS), BS, 0, stream>>>(xw, nrm, row, col, agg, E, F2);
    k_relu    <<<gblocks((long long)N * F2, BS), BS, 0, stream>>>(agg, act, (long long)N * F2);

    // ---- layer 3: act[N x 128] * W3 -> 40, scatter, log_softmax ----
    launch_gemm(act, W3, xw, N, F2, CC, stream);
    k_init_agg<<<gblocks((long long)N * CC, BS), BS, 0, stream>>>(xw, dis, b3, agg, N, CC);
    k_scatter <<<gblocks(E * CC, BS), BS, 0, stream>>>(xw, nrm, row, col, agg, E, CC);
    k_log_softmax<<<gblocks(N, BS), BS, 0, stream>>>(agg, out, N, CC);
}
